// CoolChicEncoder_33234456937097
// MI455X (gfx1250) — matrix operand order repaired and checked by reading, not compiled, not run on hardware
//
#include <hip/hip_runtime.h>
#include <hip/hip_bf16.h>

typedef __attribute__((ext_vector_type(16))) _Float16 v16h;
typedef __attribute__((ext_vector_type(8)))  float    v8f;
typedef unsigned int v4u __attribute__((ext_vector_type(4)));
typedef int          v4i __attribute__((ext_vector_type(4)));
typedef int          v8i __attribute__((ext_vector_type(8)));

// ---------------- problem constants (compile-time, from reference) ----------
#define NG 7
#define HW0 921600            // 720*1280
#define TOTAL 1228760         // sum of all grid areas
#define PADTOT 1260968        // sum of (h+8)*(w+8)
#define GAIN 16.0f
#define T_INV (1.0f/0.3f)
#define LOG_SCALE_CLAMP 13.8155f
#define RATE_OFF (6*HW0)
#define BPD_IDX (6*HW0 + TOTAL)

__constant__ int GOFF[8] = {0, 921600, 1152000, 1209600, 1224000, 1227600, 1228520, 1228760};
__constant__ int GW[7]   = {1280, 640, 320, 160, 80, 40, 20};
__constant__ int POFF[7] = {0, 937664, 1176128, 1237792, 1254256, 1258920, 1260408};
// 24 causal context offsets (row, col) inside the padded (pad=4) window
__constant__ int CR[24] = {0,1,1,1,1,1,2,2,2,2,2,2,2,3,3,3,3,3,3,3,4,4,4,4};
__constant__ int CC[24] = {4,2,3,4,5,6,1,2,3,4,5,6,7,1,2,3,4,5,6,7,0,1,2,3};

// ---------------- TDM: 1-D span global->LDS via tensor_load_to_lds ----------
// D# per CDNA5 ISA:
//   g0: [1:0]=count=1, [63:32]=lds_addr, [120:64]=global_addr, [127:126]=type=2
//   g1: [17:16]=data_size=2 (4B), [79:48]=tensor_dim0=N, [111:80]=tensor_dim1=1,
//       [127:112]=tile_dim0=N, [143:128]=tile_dim1=1, [207:160]=stride0=N
// groups 2/3 zero (tensor <= 2D). Issued once per wave; tracked by TENSORcnt.
__device__ inline void tdm_load_1d(const float* gsrc, float* ldst, int nelem) {
  unsigned lds = (unsigned)(unsigned long long)(uintptr_t)ldst; // flat[31:0] == LDS byte offset
  unsigned long long ga = (unsigned long long)(uintptr_t)gsrc;
  unsigned n = (unsigned)nelem;
  v4u g0 = {1u, lds, (unsigned)ga, (unsigned)(ga >> 32) | 0x80000000u};
  v8i g1 = {(int)(2u << 16),                 // data_size = 4 bytes
            (int)((n & 0xFFFFu) << 16),      // tensor_dim0[15:0]
            (int)((n >> 16) | (1u << 16)),   // tensor_dim0[31:16], tensor_dim1[15:0]=1
            (int)((n & 0xFFFFu) << 16),      // tile_dim0 = n
            1,                               // tile_dim1 = 1, tile_dim2 = 0
            (int)n,                          // tensor_dim0_stride[31:0]
            0, 0};
  v4i z4 = {0, 0, 0, 0};
#if __clang_major__ >= 23
  v8i z8 = {0, 0, 0, 0, 0, 0, 0, 0};
  __builtin_amdgcn_tensor_load_to_lds(g0, g1, z4, z4, z8, 0);
#else
  __builtin_amdgcn_tensor_load_to_lds(g0, g1, z4, z4, 0);
#endif
}

// ---------------- WMMA fragment helpers (wave32, 16x16x32 f16) --------------
// A fragment: 16x32 f16. lane: M = lane&15; hi half selects K interleave.
__device__ inline v16h load_a_frag(const _Float16* base, int rowStride, int kofs, int lane) {
  int M  = lane & 15;
  int hi = (lane >> 4) & 1;
  const _Float16* row = base + M * rowStride + kofs;
  v16h a;
#pragma unroll
  for (int v = 0; v < 8; ++v) {
    int K = (v < 4) ? ((hi ? 8 : 0) + 2 * v) : ((hi ? 24 : 16) + 2 * (v - 4));
    a[2 * v]     = row[K];
    a[2 * v + 1] = row[K + 1];
  }
  return a;
}
// B fragment: 32x16 f16 from f32 weights, generic strides W[k*sK + n*sN].
__device__ inline v16h load_b_frag(const float* w, int sK, int sN, int Kdim, int Ndim,
                                   int nbase, int kofs, int lane) {
  int n  = (lane & 15) + nbase;
  int hi = (lane >> 4) & 1;
  v16h b;
#pragma unroll
  for (int v = 0; v < 8; ++v) {
    int K = ((v < 4) ? (hi * 8 + 2 * v) : (16 + hi * 8 + 2 * (v - 4))) + kofs;
    float f0 = (K     < Kdim && n < Ndim) ? w[K * sK + n * sN]       : 0.f;
    float f1 = (K + 1 < Kdim && n < Ndim) ? w[(K + 1) * sK + n * sN] : 0.f;
    b[2 * v]     = (_Float16)f0;
    b[2 * v + 1] = (_Float16)f1;
  }
  return b;
}
__device__ inline v8f wmma16(v16h a, v16h b, v8f c) {
  return __builtin_amdgcn_wmma_f32_16x16x32_f16(false, a, false, b, (short)0, c, false, false);
}
__device__ inline v8f splat8(float x) {
  v8f c;
#pragma unroll
  for (int r = 0; r < 8; ++r) c[r] = x;
  return c;
}

// ---------------- kernel 1: quantize (softround) ----------------------------
__global__ void k_quantize(const float* l0, const float* l1, const float* l2,
                           const float* l3, const float* l4, const float* l5,
                           const float* l6, float* dec, float* pad) {
  int g = blockIdx.x * blockDim.x + threadIdx.x;
  if (g >= TOTAL) return;
  int gi = 0;
  while (gi < 6 && g >= GOFF[gi + 1]) ++gi;
  int loc = g - GOFF[gi];
  const float* lat = (gi == 0) ? l0 : (gi == 1) ? l1 : (gi == 2) ? l2 :
                     (gi == 3) ? l3 : (gi == 4) ? l4 : (gi == 5) ? l5 : l6;
  float v = lat[loc] * GAIN;
  float fl = floorf(v);
  float delta = v - fl - 0.5f;
  const float inv = 1.0f / tanhf(0.5f * T_INV);
  float q = fl + 0.5f * tanhf(delta * T_INV) * inv + 0.5f;
  dec[g] = q;
  int w = GW[gi];
  int y = loc / w, x = loc - y * w;
  pad[POFF[gi] + (y + 4) * (w + 8) + (x + 4)] = q;
}

// ---------------- kernel 2: ARM MLP via WMMA + Laplace rate -----------------
__global__ void __launch_bounds__(256)
k_arm(const float* pad, const float* dec,
      const float* w0, const float* b0, const float* w1, const float* b1,
      const float* wo, const float* bo, float* rate_out, float* bpd_acc) {
  __shared__ _Float16 ctxs[128 * 32];
  __shared__ _Float16 hbuf[128 * 32];
  __shared__ float outb[128 * 2];
  __shared__ float wlds[576 + 576 + 64];   // w0, w1, wout staged by TDM
  __shared__ float ssum;
  const int tid = threadIdx.x, lane = tid & 31, wav = tid >> 5;
  const int ncol = lane & 15, mbase = wav * 16 + ((lane >> 4) << 3);
  const int base = blockIdx.x * 128;

  // DMA weight matrices into LDS via the Tensor Data Mover (one wave issues)
  if (wav == 0) {
    tdm_load_1d(w0, wlds, 576);
    tdm_load_1d(w1, wlds + 576, 576);
    tdm_load_1d(wo, wlds + 1152, 48);
    __builtin_amdgcn_s_wait_tensorcnt(0);
  }

  // gather 24 context values per row into LDS (pad K to 32 with zeros)
  for (int idx = tid; idx < 128 * 32; idx += 256) {
    int row = idx >> 5, k = idx & 31;
    float v = 0.f;
    int g = base + row;
    if (k < 24 && g < TOTAL) {
      int gi = 0;
      while (gi < 6 && g >= GOFF[gi + 1]) ++gi;
      int loc = g - GOFF[gi];
      int w = GW[gi];
      int y = loc / w, x = loc - y * w;
      v = pad[POFF[gi] + (y + CR[k]) * (w + 8) + (x + CC[k])];
    }
    ctxs[idx] = (_Float16)v;
  }
  if (tid == 0) ssum = 0.f;
  __syncthreads();

  // weight fragments from LDS (arm_w* are [K][N] row-major)
  v16h B0a = load_b_frag(wlds, 24, 1, 24, 24, 0, 0, lane);
  v16h B0b = load_b_frag(wlds, 24, 1, 24, 24, 16, 0, lane);
  v16h B1a = load_b_frag(wlds + 576, 24, 1, 24, 24, 0, 0, lane);
  v16h B1b = load_b_frag(wlds + 576, 24, 1, 24, 24, 16, 0, lane);
  v16h Bo  = load_b_frag(wlds + 1152, 2, 1, 24, 2, 0, 0, lane);
  float bi0a = b0[ncol];
  float bi0b = (ncol + 16 < 24) ? b0[ncol + 16] : 0.f;
  float bi1a = b1[ncol];
  float bi1b = (ncol + 16 < 24) ? b1[ncol + 16] : 0.f;
  float bio  = (ncol < 2) ? bo[ncol] : 0.f;

  // layer 0: [16,24] @ [24,24] + b0, relu
  {
    v16h a = load_a_frag(ctxs + wav * 16 * 32, 32, 0, lane);
    v8f c0 = splat8(bi0a), c1 = splat8(bi0b);
    c0 = wmma16(a, B0a, c0);
    c1 = wmma16(a, B0b, c1);
#pragma unroll
    for (int r = 0; r < 8; ++r) {
      hbuf[(mbase + r) * 32 + ncol]      = (_Float16)fmaxf(c0[r], 0.f);
      hbuf[(mbase + r) * 32 + 16 + ncol] = (_Float16)fmaxf(c1[r], 0.f);
    }
  }
  __syncthreads();
  // layer 1: @ [24,24] + b1, relu  (reuse ctxs as output buffer)
  {
    v16h a = load_a_frag(hbuf + wav * 16 * 32, 32, 0, lane);
    v8f c0 = splat8(bi1a), c1 = splat8(bi1b);
    c0 = wmma16(a, B1a, c0);
    c1 = wmma16(a, B1b, c1);
#pragma unroll
    for (int r = 0; r < 8; ++r) {
      ctxs[(mbase + r) * 32 + ncol]      = (_Float16)fmaxf(c0[r], 0.f);
      ctxs[(mbase + r) * 32 + 16 + ncol] = (_Float16)fmaxf(c1[r], 0.f);
    }
  }
  __syncthreads();
  // output layer: @ [24,2] + bo
  {
    v16h a = load_a_frag(ctxs + wav * 16 * 32, 32, 0, lane);
    v8f c = splat8(bio);
    c = wmma16(a, Bo, c);
    if (ncol < 2) {
#pragma unroll
      for (int r = 0; r < 8; ++r) outb[(mbase + r) * 2 + ncol] = c[r];
    }
  }
  __syncthreads();

  // Laplace rate per row
  if (tid < 128) {
    int g = base + tid;
    if (g < TOTAL) {
      float mu = outb[tid * 2 + 0];
      float sl = fminf(fmaxf(outb[tid * 2 + 1], -LOG_SCALE_CLAMP), LOG_SCALE_CLAMP);
      float scale = expf(-0.5f * sl);
      float xv = dec[g];
      float dH = xv + 0.5f - mu, dL = xv - 0.5f - mu;
      float sH = (dH > 0.f) ? 1.f : ((dH < 0.f) ? -1.f : 0.f);
      float sL = (dL > 0.f) ? 1.f : ((dL < 0.f) ? -1.f : 0.f);
      float cH = 0.5f - 0.5f * sH * expm1f(-fabsf(dH) / scale);
      float cL = 0.5f - 0.5f * sL * expm1f(-fabsf(dL) / scale);
      float p = fmaxf(cH - cL, 1.52587890625e-5f); // 2^-16
      float rate = -log2f(p);
      rate_out[g] = rate;
      atomicAdd(&ssum, rate);
    }
  }
  __syncthreads();
  if (tid == 0) atomicAdd(bpd_acc, ssum);
}

__global__ void k_bpd(const float* acc, float* out) {
  out[BPD_IDX] = acc[0] * (1.0f / (720.0f * 1280.0f * 3.0f));
}

// ---------------- upsampling chain ------------------------------------------
__global__ void k_pre7(const float* in, float* out, int C, int H, int W, const float* k7) {
  int t = blockIdx.x * blockDim.x + threadIdx.x;
  int n = C * H * W;
  if (t >= n) return;
  int ch = t / (H * W), rem = t - ch * H * W;
  int y = rem / W, x = rem - y * W;
  const float* p = in + ch * H * W;
  float acc = 0.f;
#pragma unroll
  for (int ky = 0; ky < 7; ++ky) {
    int yy = y + ky - 3;
    if (yy < 0 || yy >= H) continue;
#pragma unroll
    for (int kx = 0; kx < 7; ++kx) {
      int xx = x + kx - 3;
      if (xx < 0 || xx >= W) continue;
      acc += p[yy * W + xx] * k7[ky * 7 + kx];
    }
  }
  out[t] = acc;
}

// transposed conv 8x8 stride 2 (VALID) + crop [3:3+Hout, 3:3+Wout];
// writes channels 1..C of out (channel 0 is reserved for the latent concat).
__global__ void k_convT8(const float* in, float* out, int C, int Hin, int Win,
                         int Hout, int Wout, const float* k8) {
  int t = blockIdx.x * blockDim.x + threadIdx.x;
  int n = C * Hout * Wout;
  if (t >= n) return;
  int ch = t / (Hout * Wout), rem = t - ch * Hout * Wout;
  int y = rem / Wout, x = rem - y * Wout;
  int yo = y + 3, xo = x + 3;
  const float* p = in + ch * Hin * Win;
  float acc = 0.f;
#pragma unroll
  for (int j = 0; j < 4; ++j) {
    int ky = (yo & 1) + 2 * j;
    int yi = (yo - ky) >> 1;
    if (yi < 0 || yi >= Hin) continue;
#pragma unroll
    for (int i = 0; i < 4; ++i) {
      int kx = (xo & 1) + 2 * i;
      int xi = (xo - kx) >> 1;
      if (xi < 0 || xi >= Win) continue;
      acc += p[yi * Win + xi] * k8[ky * 8 + kx];
    }
  }
  out[(ch + 1) * Hout * Wout + y * Wout + x] = acc;
}

__global__ void k_copy(const float* src, float* dst, int n) {
  int t = blockIdx.x * blockDim.x + threadIdx.x;
  if (t < n) dst[t] = src[t];
}

// ---------------- fused synthesis 1x1 convs (7->48 relu ->6) via WMMA -------
__global__ void __launch_bounds__(256)
k_syn01(const float* x, const float* w0, const float* b0,
        const float* w1, const float* b1, float* s1) {
  __shared__ _Float16 inA[128 * 32];
  __shared__ _Float16 hbuf[128 * 64];
  __shared__ float wlds[336 + 288];        // syn_w0, syn_w1 staged by TDM
  const int tid = threadIdx.x, lane = tid & 31, wav = tid >> 5;
  const int ncol = lane & 15, mbase = wav * 16 + ((lane >> 4) << 3);
  const int base = blockIdx.x * 128;

  if (wav == 0) {
    tdm_load_1d(w0, wlds, 336);
    tdm_load_1d(w1, wlds + 336, 288);
    __builtin_amdgcn_s_wait_tensorcnt(0);
  }

  // gather 7 input channels per pixel (pad K to 32); zero hbuf K-pad cols 48..63
  for (int idx = tid; idx < 128 * 32; idx += 256) {
    int row = idx >> 5, k = idx & 31;
    float v = 0.f;
    int g = base + row;
    if (k < 7 && g < HW0) v = x[k * HW0 + g];
    inA[idx] = (_Float16)v;
  }
  for (int idx = tid; idx < 128 * 16; idx += 256) {
    int row = idx >> 4, k = idx & 15;
    hbuf[row * 64 + 48 + k] = (_Float16)0.f;
  }
  __syncthreads();

  // syn_w0 is [48][7] ([N][K]); syn_w1 is [6][48] ([N][K]) -- read from LDS
  v16h B0a = load_b_frag(wlds, 1, 7, 7, 48, 0, 0, lane);
  v16h B0b = load_b_frag(wlds, 1, 7, 7, 48, 16, 0, lane);
  v16h B0c = load_b_frag(wlds, 1, 7, 7, 48, 32, 0, lane);
  v16h B1a = load_b_frag(wlds + 336, 1, 48, 48, 6, 0, 0, lane);
  v16h B1b = load_b_frag(wlds + 336, 1, 48, 48, 6, 0, 32, lane);
  float bi0a = b0[ncol], bi0b = b0[ncol + 16], bi0c = b0[ncol + 32];
  float bi1  = (ncol < 6) ? b1[ncol] : 0.f;

  // layer 0: [16,7] @ [7,48] + b0, relu
  {
    v16h a = load_a_frag(inA + wav * 16 * 32, 32, 0, lane);
    v8f c0 = splat8(bi0a), c1 = splat8(bi0b), c2 = splat8(bi0c);
    c0 = wmma16(a, B0a, c0);
    c1 = wmma16(a, B0b, c1);
    c2 = wmma16(a, B0c, c2);
#pragma unroll
    for (int r = 0; r < 8; ++r) {
      hbuf[(mbase + r) * 64 + ncol]      = (_Float16)fmaxf(c0[r], 0.f);
      hbuf[(mbase + r) * 64 + 16 + ncol] = (_Float16)fmaxf(c1[r], 0.f);
      hbuf[(mbase + r) * 64 + 32 + ncol] = (_Float16)fmaxf(c2[r], 0.f);
    }
  }
  __syncthreads();
  // layer 1: [16,48] @ [48,6] + b1 (two K tiles accumulate)
  {
    v16h alo = load_a_frag(hbuf + wav * 16 * 64, 64, 0, lane);
    v16h ahi = load_a_frag(hbuf + wav * 16 * 64, 64, 32, lane);
    v8f c = splat8(bi1);
    c = wmma16(alo, B1a, c);
    c = wmma16(ahi, B1b, c);
    if (ncol < 6) {
#pragma unroll
      for (int r = 0; r < 8; ++r) {
        int g = base + mbase + r;
        if (g < HW0) s1[ncol * HW0 + g] = c[r];
      }
    }
  }
}

// ---------------- 3x3 residual convs ----------------------------------------
__global__ void k_conv3res_relu(const float* s1, const float* w2, const float* b2, float* s2) {
  int t = blockIdx.x * blockDim.x + threadIdx.x;
  if (t >= HW0) return;
  int y = t / 1280, x = t - y * 1280;
  float acc[6];
#pragma unroll
  for (int o = 0; o < 6; ++o) acc[o] = b2[o];
#pragma unroll
  for (int ky = 0; ky < 3; ++ky) {
    int yy = y + ky - 1;
    if (yy < 0 || yy >= 720) continue;
#pragma unroll
    for (int kx = 0; kx < 3; ++kx) {
      int xx = x + kx - 1;
      if (xx < 0 || xx >= 1280) continue;
#pragma unroll
      for (int i = 0; i < 6; ++i) {
        float v = s1[i * HW0 + yy * 1280 + xx];
#pragma unroll
        for (int o = 0; o < 6; ++o)
          acc[o] += v * w2[((o * 6 + i) * 3 + ky) * 3 + kx];
      }
    }
  }
#pragma unroll
  for (int o = 0; o < 6; ++o)
    s2[o * HW0 + t] = fmaxf(acc[o] + s1[o * HW0 + t], 0.f);
}

__global__ void k_syn3_out(const float* s2, const float* w3, const float* b3, float* out) {
  int t = blockIdx.x * blockDim.x + threadIdx.x;
  if (t >= HW0) return;
  int y = t / 1280, x = t - y * 1280;
  float acc[6];
#pragma unroll
  for (int o = 0; o < 6; ++o) acc[o] = b3[o];
#pragma unroll
  for (int ky = 0; ky < 3; ++ky) {
    int yy = y + ky - 1;
    if (yy < 0 || yy >= 720) continue;
#pragma unroll
    for (int kx = 0; kx < 3; ++kx) {
      int xx = x + kx - 1;
      if (xx < 0 || xx >= 1280) continue;
#pragma unroll
      for (int i = 0; i < 6; ++i) {
        float v = s2[i * HW0 + yy * 1280 + xx];
#pragma unroll
        for (int o = 0; o < 6; ++o)
          acc[o] += v * w3[((o * 6 + i) * 3 + ky) * 3 + kx];
      }
    }
  }
#pragma unroll
  for (int o = 0; o < 3; ++o)
    out[o * HW0 + t] = acc[o] + s2[o * HW0 + t];        // mu
#pragma unroll
  for (int o = 0; o < 3; ++o) {
    float v = acc[3 + o] + s2[(3 + o) * HW0 + t];
    v = fminf(fmaxf(v, -LOG_SCALE_CLAMP), LOG_SCALE_CLAMP);
    out[(3 + o) * HW0 + t] = expf(-0.5f * v);           // scale
  }
}

// ---------------- host driver -----------------------------------------------
// workspace layout (floats)
#define WS_BPD  ((size_t)0)
#define WS_DEC  ((size_t)16)
#define WS_PAD  (WS_DEC + TOTAL)          // 1228776
#define WS_BUFA (WS_PAD + PADTOT)         // 2489744
#define WS_BUFB (WS_BUFA + (size_t)7*HW0) // 8940944
#define WS_BUFT (WS_BUFB + (size_t)7*HW0) // 15392144 (+6*230400 -> ~67 MB total)

extern "C" void kernel_launch(void* const* d_in, const int* in_sizes, int n_in,
                              void* d_out, int out_size, void* d_ws, size_t ws_size,
                              hipStream_t stream) {
  const float* lat[7];
  for (int i = 0; i < 7; ++i) lat[i] = (const float*)d_in[i];
  const float* aw0 = (const float*)d_in[7];
  const float* ab0 = (const float*)d_in[8];
  const float* aw1 = (const float*)d_in[9];
  const float* ab1 = (const float*)d_in[10];
  const float* awo = (const float*)d_in[11];
  const float* abo = (const float*)d_in[12];
  const float* prek = (const float*)d_in[13];
  const float* upsk = (const float*)d_in[14];
  const float* sw0 = (const float*)d_in[15];
  const float* sb0 = (const float*)d_in[16];
  const float* sw1 = (const float*)d_in[17];
  const float* sb1 = (const float*)d_in[18];
  const float* sw2 = (const float*)d_in[19];
  const float* sb2 = (const float*)d_in[20];
  const float* sw3 = (const float*)d_in[21];
  const float* sb3 = (const float*)d_in[22];

  float* ws   = (float*)d_ws;
  float* bpd  = ws + WS_BPD;
  float* dec  = ws + WS_DEC;
  float* pad  = ws + WS_PAD;
  float* bufA = ws + WS_BUFA;
  float* bufB = ws + WS_BUFB;
  float* bufT = ws + WS_BUFT;
  float* out  = (float*)d_out;

  hipMemsetAsync(pad, 0, (size_t)PADTOT * sizeof(float), stream);
  hipMemsetAsync(bpd, 0, sizeof(float), stream);

  k_quantize<<<(TOTAL + 255) / 256, 256, 0, stream>>>(
      lat[0], lat[1], lat[2], lat[3], lat[4], lat[5], lat[6], dec, pad);

  k_arm<<<(TOTAL + 127) / 128, 256, 0, stream>>>(
      pad, dec, aw0, ab0, aw1, ab1, awo, abo, out + RATE_OFF, bpd);
  k_bpd<<<1, 1, 0, stream>>>(bpd, out);

  // upsampling chain: dec[6] -> ... -> 7ch @ 720x1280
  static const int SH[7] = {720, 360, 180, 90, 45, 23, 12};
  static const int SW[7] = {1280, 640, 320, 160, 80, 40, 20};
  static const long GOh[7] = {0, 921600, 1152000, 1209600, 1224000, 1227600, 1228520};
  const float* xcur = dec + GOh[6];
  int C = 1;
  for (int step = 0; step < 6; ++step) {
    int i = 5 - step;
    int hin = SH[i + 1], win = SW[i + 1], ht = SH[i], wt = SW[i];
    float* xnext = (step & 1) ? bufB : bufA;
    const float* k7 = prek + step * 49;
    const float* k8 = upsk + step * 64;
    int n1 = C * hin * win;
    k_pre7<<<(n1 + 255) / 256, 256, 0, stream>>>(xcur, bufT, C, hin, win, k7);
    int n2 = C * ht * wt;
    k_convT8<<<(n2 + 255) / 256, 256, 0, stream>>>(bufT, xnext, C, hin, win, ht, wt, k8);
    k_copy<<<(ht * wt + 255) / 256, 256, 0, stream>>>(dec + GOh[i], xnext, ht * wt);
    xcur = xnext;
    ++C;
  }
  // xcur == bufB (7 channels, full res)
  k_syn01<<<HW0 / 128, 256, 0, stream>>>(xcur, sw0, sb0, sw1, sb1, bufA);
  k_conv3res_relu<<<(HW0 + 255) / 256, 256, 0, stream>>>(bufA, sw2, sb2, bufB);
  k_syn3_out<<<(HW0 + 255) / 256, 256, 0, stream>>>(bufB, sw3, sb3, out);
}
